// MRIL_37726992728296
// MI455X (gfx1250) — compile-verified, matrix operand order checked by reading, not run on hardware
//
#include <hip/hip_runtime.h>

typedef __attribute__((ext_vector_type(16))) __bf16 v16bf;
typedef __attribute__((ext_vector_type(8)))  float  v8f;

#define NN   8192
#define HD   128
#define KN   8
#define ROWU 64            // u32 words per bf16 row of 128
#define EPSR 1e-6f
#define BALL (1.0f - 1e-4f)

struct F4 { float x, y, z, w; };

__device__ __forceinline__ F4 ld4(const float* p){ F4 r{p[0],p[1],p[2],p[3]}; return r; }
__device__ __forceinline__ void st4(float* p, F4 v){ p[0]=v.x; p[1]=v.y; p[2]=v.z; p[3]=v.w; }
__device__ __forceinline__ F4 vadd(F4 a, F4 b){ return {a.x+b.x, a.y+b.y, a.z+b.z, a.w+b.w}; }
__device__ __forceinline__ F4 vsub(F4 a, F4 b){ return {a.x-b.x, a.y-b.y, a.z-b.z, a.w-b.w}; }
__device__ __forceinline__ F4 vmul(F4 a, F4 b){ return {a.x*b.x, a.y*b.y, a.z*b.z, a.w*b.w}; }
__device__ __forceinline__ F4 vscale(F4 a, float s){ return {a.x*s, a.y*s, a.z*s, a.w*s}; }
__device__ __forceinline__ float pdot(F4 a, F4 b){ return a.x*b.x + a.y*b.y + a.z*b.z + a.w*b.w; }

// wave32 full-lane reduction (all lanes get the sum)
__device__ __forceinline__ float wsum(float v){
  v += __shfl_xor(v, 16, 32);
  v += __shfl_xor(v,  8, 32);
  v += __shfl_xor(v,  4, 32);
  v += __shfl_xor(v,  2, 32);
  v += __shfl_xor(v,  1, 32);
  return v;
}
__device__ __forceinline__ float vdotw(F4 a, F4 b){ return wsum(pdot(a, b)); }
__device__ __forceinline__ float vnorm(F4 a){ return sqrtf(vdotw(a, a) + 1e-15f); }

__device__ __forceinline__ float tank(float u, bool sph){
  if (sph){ u = fminf(fmaxf(u, -1.47079f), 1.47079f); return tanf(u); }
  return tanhf(u);
}
__device__ __forceinline__ float artank(float u, bool sph){
  if (sph) return atanf(u);
  u = fminf(fmaxf(u, -1.0f + 1e-5f), 1.0f - 1e-5f);
  return atanhf(u);
}
__device__ __forceinline__ float safden(float d){
  return (d >= 0.0f) ? fmaxf(d, EPSR) : fminf(d, -EPSR);
}

__device__ __forceinline__ F4 projk(F4 v, bool sph){
  if (sph) return v;
  float n = vnorm(v);
  if (n > BALL) v = vscale(v, BALL / n);
  return v;
}
__device__ __forceinline__ F4 expmap0(F4 u, bool sph){
  float n = vnorm(u);
  F4 r = vscale(u, tank(n, sph) / n);
  return projk(r, sph);
}
__device__ __forceinline__ F4 logmap0(F4 y, bool sph){
  float n = vnorm(y);
  return vscale(y, artank(n, sph) / n);
}
__device__ __forceinline__ F4 mobius_add(F4 x, F4 y, bool sph){
  float kf = sph ? 1.0f : -1.0f;
  float x2 = vdotw(x, x), y2 = vdotw(y, y), xy = vdotw(x, y);
  float A = 1.0f - 2.0f*kf*xy - kf*y2;
  float B = 1.0f + kf*x2;
  float den = safden(1.0f - 2.0f*kf*xy + kf*kf*x2*y2);
  F4 r = vadd(vscale(x, A/den), vscale(y, B/den));
  return projk(r, sph);
}
__device__ __forceinline__ F4 mpm(F4 w, F4 x, bool sph){          // mobius_pw_mul
  F4 wx = vmul(w, x);
  float nx  = vnorm(x);
  float nwx = vnorm(wx);
  float t = tank(nwx/nx * artank(nx, sph), sph);
  return projk(vscale(wx, t/nwx), sph);
}
__device__ __forceinline__ F4 mpms(F4 w, float g, bool sph){      // x = broadcast scalar g
  F4 wx = vscale(w, g);
  float nx  = sqrtf(g*g + 1e-15f);
  float nwx = vnorm(wx);
  float t = tank(nwx/nx * artank(nx, sph), sph);
  return projk(vscale(wx, t/nwx), sph);
}
__device__ __forceinline__ F4 msmul(float r, F4 x, bool sph){     // mobius_smul
  float n = vnorm(x);
  float t = tank(r * artank(n, sph), sph);
  return projk(vscale(x, t/n), sph);
}
__device__ __forceinline__ F4 sigmoid4(F4 v){
  return {1.0f/(1.0f+expf(-v.x)), 1.0f/(1.0f+expf(-v.y)),
          1.0f/(1.0f+expf(-v.z)), 1.0f/(1.0f+expf(-v.w))};
}
__device__ __forceinline__ F4 tanh4(F4 v){
  return {tanhf(v.x), tanhf(v.y), tanhf(v.z), tanhf(v.w)};
}
// 3-way weighted midpoint (always Poincare k=-1 per reference), scaled by 3
__device__ __forceinline__ F4 wmid3x3(F4 a, F4 b, F4 c){
  float la = 2.0f/(1.0f - vdotw(a, a));
  float lb = 2.0f/(1.0f - vdotw(b, b));
  float lc = 2.0f/(1.0f - vdotw(c, c));
  F4 num = vadd(vadd(vscale(a, la), vscale(b, lb)), vscale(c, lc));
  float den = fmaxf(fabsf((la-1.0f)+(lb-1.0f)+(lc-1.0f)), EPSR);
  F4 m = msmul(0.5f, vscale(num, 1.0f/den), false);
  return vscale(m, 3.0f);
}

// ---- bf16 packing ----
__device__ __forceinline__ unsigned short f2bf(float f){
  unsigned u = __float_as_uint(f);
  unsigned r = u + 0x7FFFu + ((u >> 16) & 1u);   // RNE
  return (unsigned short)(r >> 16);
}
__device__ __forceinline__ unsigned pk2(float a, float b){
  return (unsigned)f2bf(a) | ((unsigned)f2bf(b) << 16);
}
__device__ __forceinline__ void stbf(unsigned* row, int lane, F4 v){
  row[lane*2]   = pk2(v.x, v.y);
  row[lane*2+1] = pk2(v.z, v.w);
}

// ====================== weight packing ======================
__global__ __launch_bounds__(256) void k_pack(const float* __restrict__ src,
                                              unsigned* __restrict__ dst, int npairs){
  int i = blockIdx.x*blockDim.x + threadIdx.x;
  if (i < npairs) dst[i] = pk2(src[2*i], src[2*i+1]);
}

// ====================== WMMA GEMM: Y[M,Ho] = A[M,128] @ W[Ho,128]^T + b ======================
union AFrag { unsigned u[8]; v16bf b; };

__global__ __launch_bounds__(256) void k_gemm(const unsigned* __restrict__ A,
                                              const unsigned* __restrict__ W,
                                              const float* __restrict__ bias,
                                              float* __restrict__ Y, int Ho){
  int lane = threadIdx.x & 31, wv = threadIdx.x >> 5;
  int half = lane >> 4, mr = lane & 15;
  int row0 = blockIdx.x * 16;
  const unsigned* Arow = A + (size_t)(row0 + mr) * ROWU;
  int tiles = Ho >> 4;
  for (int tn = wv; tn < tiles; tn += 8){
    const unsigned* Wrow = W + (size_t)(tn*16 + mr) * ROWU;
    if (tn + 8 < tiles)
      __builtin_prefetch(W + (size_t)((tn+8)*16 + mr) * ROWU, 0, 3);
    v8f acc;
    #pragma unroll
    for (int i = 0; i < 8; i++) acc[i] = 0.0f;
    #pragma unroll
    for (int kc = 0; kc < 4; kc++){
      AFrag fa, fb;
      #pragma unroll
      for (int i = 0; i < 8; i++){
        // 16-bit A-matrix 16x32 lane layout: VGPR i holds K-pair p (ISA 7.12.2)
        int p = (i < 4) ? (i + 4*half) : (i - 4 + 4*half + 8);
        fa.u[i] = Arow[kc*16 + p];
        fb.u[i] = Wrow[kc*16 + p];
      }
      acc = __builtin_amdgcn_wmma_f32_16x16x32_bf16(false, fa.b, false, fb.b,
                                                    (short)0, acc, false, false);
    }
    float bv = bias[tn*16 + mr];
    #pragma unroll
    for (int r = 0; r < 8; r++)
      Y[(size_t)(row0 + r + half*8) * Ho + tn*16 + mr] = acc[r] + bv;
  }
}

// ====================== pre-op: build bf16 GEMM inputs ======================
__global__ __launch_bounds__(256) void k_preop(
    const float* __restrict__ x,  const float* __restrict__ h1,
    const float* __restrict__ h2, const float* __restrict__ h3,
    const float* __restrict__ c1, const float* __restrict__ c2,
    const float* __restrict__ c3,
    unsigned* bt0p, unsigned* bt0s, unsigned* bte,
    unsigned* bl1,  unsigned* bl2,  unsigned* bh3,
    unsigned* blc1, unsigned* blc2, unsigned* bc3){
  int lane = threadIdx.x & 31, wv = threadIdx.x >> 5;
  int j = blockIdx.x * 8 + wv;
  int o = j*HD + lane*4, ob = j*ROWU;
  F4 xv = ld4(x + o);
  stbf(bt0p + ob, lane, logmap0(expmap0(xv, false), false));
  stbf(bt0s + ob, lane, logmap0(expmap0(xv, true),  true));
  stbf(bte  + ob, lane, xv);
  F4 h1v = ld4(h1+o), h2v = ld4(h2+o), h3v = ld4(h3+o);
  stbf(bl1 + ob, lane, logmap0(h1v, false));
  stbf(bl2 + ob, lane, logmap0(h2v, true));
  stbf(bh3 + ob, lane, h3v);
  F4 c1v = ld4(c1+o), c2v = ld4(c2+o), c3v = ld4(c3+o);
  stbf(blc1 + ob, lane, logmap0(c1v, false));
  stbf(blc2 + ob, lane, logmap0(c2v, true));
  stbf(bc3  + ob, lane, c3v);
}

// ====================== per-node epilogue after the 12 GEMMs ======================
__global__ __launch_bounds__(256) void k_post1(
    float* q_p, float* q_s,                       // in: gemm out; out: x_1p / x_1s (in place)
    float* up1, float* up2, float* up3,           // in: gemm; out: H1P, H1S, H1E
    float* uf1, float* uf2, float* uf3,           // in: gemm; out: FCKT_P, FCKT_S, FCKT_E
    const float* wc1, const float* wc2, const float* wc3,
    const float* h1, const float* h2, const float* h3,
    const float* c1, const float* c2, const float* c3,
    const float* __restrict__ del_t, const float* __restrict__ dsc,
    float* y2h1p, float* y2h1s, float* y2fp, float* y2fs){
  int lane = threadIdx.x & 31, wv = threadIdx.x >> 5;
  int j = blockIdx.x * 8 + wv;
  int o = j*HD + lane*4;
  F4 qp = ld4(q_p+o), qs = ld4(q_s+o);
  F4 u1 = ld4(up1+o), u2 = ld4(up2+o), u3 = ld4(up3+o);
  F4 f1 = ld4(uf1+o), f2 = ld4(uf2+o), f3 = ld4(uf3+o);
  F4 w1 = ld4(wc1+o), w2 = ld4(wc2+o), w3 = ld4(wc3+o);
  F4 h1v = ld4(h1+o), h2v = ld4(h2+o), h3v = ld4(h3+o);
  F4 c1v = ld4(c1+o), c2v = ld4(c2+o), c3v = ld4(c3+o);
  float g = dsc[0] / (del_t[j] + 1.0f);

  st4(q_p+o, expmap0(qp, false));   // x_1p
  st4(q_s+o, expmap0(qs, true));    // x_1s   (x_1e = q_e unchanged)

  F4 S1 = logmap0(sigmoid4(expmap0(u1, false)), false);
  F4 S2 = logmap0(sigmoid4(expmap0(u2, true)),  true);
  F4 S3 = sigmoid4(u3);

  F4 ht1p = mpm(S1, h1v, false);
  F4 ht2p = mpm(S2, expmap0(logmap0(h2v, true),  false), false);
  F4 ht3p = mpm(S3, expmap0(h3v, false), false);
  F4 H1P  = wmid3x3(ht1p, ht2p, ht3p);

  F4 ht1s = mpm(S1, expmap0(logmap0(h1v, false), true), false);
  F4 ht2s = mpm(S2, h2v, true);
  F4 ht3s = mpm(S3, expmap0(h3v, true), true);
  F4 H1S  = wmid3x3(ht1s, ht2s, ht3s);

  F4 H1E = vadd(vadd(logmap0(ht1p, false), logmap0(ht2s, true)), vmul(S3, h3v));

  st4(up1+o, H1P); st4(up2+o, H1S); st4(up3+o, H1E);
  float yp = vdotw(H1P, H1P), ys = vdotw(H1S, H1S);

  // Poincare cell
  F4 cskp = expmap0(tanh4(logmap0(expmap0(w1, false), false)), false);
  F4 cktp = mobius_add(mobius_add(vscale(cskp, -1.0f), c1v, false),
                       mpms(cskp, g, false), false);
  F4 Fp  = sigmoid4(logmap0(expmap0(f1, false), false));
  F4 FCp = mpm(Fp, cktp, false);
  st4(uf1+o, FCp);
  float yfp = vdotw(FCp, FCp);

  // Sphere cell
  F4 csks = expmap0(tanh4(logmap0(expmap0(w2, true), true)), true);
  F4 ckts = mobius_add(mobius_add(vscale(csks, -1.0f), c2v, true),
                       mpms(csks, g, true), true);
  F4 Fs  = sigmoid4(logmap0(expmap0(f2, true), true));
  F4 FCs = mpm(Fs, ckts, true);
  st4(uf2+o, FCs);
  float yfs = vdotw(FCs, FCs);

  // Euclidean cell
  F4 cske = tanh4(w3);
  F4 ckte = vadd(vsub(c3v, cske), vscale(cske, g));
  F4 FCe  = vmul(sigmoid4(f3), ckte);
  st4(uf3+o, FCe);

  if (lane == 0){ y2h1p[j] = yp; y2h1s[j] = ys; y2fp[j] = yfp; y2fs[j] = yfs; }
}

// ====================== attention + cell reductions over K=8 neighbors ======================
__global__ __launch_bounds__(256) void k_attn(
    const int* __restrict__ nbr,
    const float* x1p, const float* x1s, const float* x1e,
    const float* H1P, const float* H1S, const float* H1E,
    const float* FCp, const float* FCs, const float* FCe,
    const float* y2h1p, const float* y2h1s, const float* y2fp, const float* y2fs,
    float* htp, float* hts, float* hte,
    float* cc1, float* cc2, float* cc3,
    unsigned* ltp, unsigned* lts, unsigned* lte){
  int lane = threadIdx.x & 31, wv = threadIdx.x >> 5;
  int n = blockIdx.x * 8 + wv;
  int o = n*HD + lane*4;
  F4 qp = ld4(x1p+o), qs = ld4(x1s+o), qe = ld4(x1e+o);
  float q2p = vdotw(qp, qp), q2s = vdotw(qs, qs);

  int idx[KN]; float scp[KN], scs[KN], sce[KN];
  #pragma unroll
  for (int k = 0; k < KN; k++){
    int j = nbr[n*KN + k]; idx[k] = j;
    size_t jo = (size_t)j*HD + lane*4;
    { // Poincare distance via quadratic form of mobius_add(-q, y)
      F4 yv = ld4(H1P + jo);
      float y2 = y2h1p[j];
      float xy_ = -vdotw(qp, yv);
      const float kf = -1.0f;
      float A = 1.0f - 2.0f*kf*xy_ - kf*y2;
      float B = 1.0f + kf*q2p;
      float den = safden(1.0f - 2.0f*kf*xy_ + kf*kf*q2p*y2);
      float n2 = (A*A*q2p + B*B*y2 + 2.0f*A*B*xy_) / (den*den);
      float nn = fminf(sqrtf(n2 + 1e-15f), BALL);
      scp[k] = -2.0f * artank(nn, false);
    }
    { // Sphere distance
      F4 yv = ld4(H1S + jo);
      float y2 = y2h1s[j];
      float xy_ = -vdotw(qs, yv);
      const float kf = 1.0f;
      float A = 1.0f - 2.0f*kf*xy_ - kf*y2;
      float B = 1.0f + kf*q2s;
      float den = safden(1.0f - 2.0f*kf*xy_ + kf*kf*q2s*y2);
      float n2 = (A*A*q2s + B*B*y2 + 2.0f*A*B*xy_) / (den*den);
      scs[k] = -2.0f * artank(sqrtf(n2 + 1e-15f), true);
    }
    sce[k] = vdotw(qe, ld4(H1E + jo)) * 0.0883883476f;   // 1/sqrt(128)
  }
  float mp = scp[0], ms = scs[0], me = sce[0];
  #pragma unroll
  for (int k = 1; k < KN; k++){
    mp = fmaxf(mp, scp[k]); ms = fmaxf(ms, scs[k]); me = fmaxf(me, sce[k]);
  }
  float ep[KN], es[KN], ee[KN]; float sp = 0, ss = 0, se = 0;
  #pragma unroll
  for (int k = 0; k < KN; k++){
    ep[k] = expf(scp[k]-mp); sp += ep[k];
    es[k] = expf(scs[k]-ms); ss += es[k];
    ee[k] = expf(sce[k]-me); se += ee[k];
  }
  F4 nump{0,0,0,0}, nums{0,0,0,0}, nume{0,0,0,0};
  F4 a1{0,0,0,0}, a2{0,0,0,0}, a3{0,0,0,0};
  float dp = 0, ds = 0, d1 = 0, d2 = 0;
  #pragma unroll
  for (int k = 0; k < KN; k++){
    int j = idx[k];
    size_t jo = (size_t)j*HD + lane*4;
    float ap = ep[k]/sp, as = es[k]/ss, ae = ee[k]/se;
    float lp = 2.0f/(1.0f - y2h1p[j]);       // lambda, k=-1
    nump = vadd(nump, vscale(ld4(H1P + jo), ap*lp)); dp += ap*(lp - 1.0f);
    float ls = 2.0f/(1.0f + y2h1s[j]);       // lambda, k=+1
    nums = vadd(nums, vscale(ld4(H1S + jo), as*ls)); ds += as*(ls - 1.0f);
    nume = vadd(nume, vscale(ld4(H1E + jo), ae));
    float l1 = 2.0f/(1.0f - y2fp[j]);
    a1 = vadd(a1, vscale(ld4(FCp + jo), l1)); d1 += l1 - 1.0f;
    float l2 = 2.0f/(1.0f + y2fs[j]);
    a2 = vadd(a2, vscale(ld4(FCs + jo), l2)); d2 += l2 - 1.0f;
    a3 = vadd(a3, ld4(FCe + jo));
  }
  F4 hp = msmul(0.5f, vscale(nump, 1.0f/fmaxf(fabsf(dp), EPSR)), false);
  F4 hs = msmul(0.5f, vscale(nums, 1.0f/fmaxf(fabsf(ds), EPSR)), true);
  st4(htp+o, hp); st4(hts+o, hs); st4(hte+o, nume);
  stbf(ltp + (size_t)n*ROWU, lane, logmap0(hp, false));
  stbf(lts + (size_t)n*ROWU, lane, logmap0(hs, true));
  stbf(lte + (size_t)n*ROWU, lane, nume);
  F4 c1r = msmul(0.5f, vscale(a1, 1.0f/fmaxf(fabsf(d1), EPSR)), false);
  F4 c2r = msmul(0.5f, vscale(a2, 1.0f/fmaxf(fabsf(d2), EPSR)), true);
  st4(cc1+o, c1r); st4(cc2+o, c2r); st4(cc3+o, a3);
}

// ====================== final apply (384-dim mobius add + gates) ======================
__device__ __forceinline__ void geom_apply(const float* giou, const float* iou,
                                           const float* ccv, float* oH, float* oC,
                                           int n, int lane, bool sph){
  F4 G[3], I[3];
  #pragma unroll
  for (int t = 0; t < 3; t++){
    size_t off = (size_t)n*384 + t*HD + lane*4;
    G[t] = ld4(giou + off);
    I[t] = ld4(iou + off);
  }
  // expmap0 over 384 dims
  float pg = 0.0f;
  #pragma unroll
  for (int t = 0; t < 3; t++) pg += pdot(G[t], G[t]);
  float ng = sqrtf(wsum(pg) + 1e-15f);
  float s = tank(ng, sph) / ng;
  #pragma unroll
  for (int t = 0; t < 3; t++) G[t] = vscale(G[t], s);
  if (!sph){
    pg = 0.0f;
    #pragma unroll
    for (int t = 0; t < 3; t++) pg += pdot(G[t], G[t]);
    float nn = sqrtf(wsum(pg) + 1e-15f);
    if (nn > BALL){ s = BALL/nn;
      #pragma unroll
      for (int t = 0; t < 3; t++) G[t] = vscale(G[t], s); }
  }
  // mobius_add over 384 dims
  float px = 0, py = 0, pxy = 0;
  #pragma unroll
  for (int t = 0; t < 3; t++){ px += pdot(I[t],I[t]); py += pdot(G[t],G[t]); pxy += pdot(I[t],G[t]); }
  float x2 = wsum(px), y2 = wsum(py), xy = wsum(pxy);
  float kf = sph ? 1.0f : -1.0f;
  float A = 1.0f - 2.0f*kf*xy - kf*y2;
  float B = 1.0f + kf*x2;
  float den = safden(1.0f - 2.0f*kf*xy + kf*kf*x2*y2);
  F4 R[3];
  #pragma unroll
  for (int t = 0; t < 3; t++) R[t] = vadd(vscale(I[t], A/den), vscale(G[t], B/den));
  if (!sph){
    float pr = 0;
    #pragma unroll
    for (int t = 0; t < 3; t++) pr += pdot(R[t], R[t]);
    float nn = sqrtf(wsum(pr) + 1e-15f);
    if (nn > BALL){ float sc = BALL/nn;
      #pragma unroll
      for (int t = 0; t < 3; t++) R[t] = vscale(R[t], sc); }
  }
  F4 ig = sigmoid4(logmap0(R[0], sph));
  F4 og = sigmoid4(logmap0(R[1], sph));
  F4 ug = tanh4(logmap0(R[2], sph));
  F4 cv = ld4(ccv + (size_t)n*HD + lane*4);
  F4 nc = mobius_add(mpm(ig, ug, sph), cv, sph);
  F4 nh = mpm(og, tanh4(logmap0(nc, sph)), sph);
  st4(oH + (size_t)n*HD + lane*4, nh);
  st4(oC + (size_t)n*HD + lane*4, nc);
}

__global__ __launch_bounds__(256) void k_apply(
    const float* g1, const float* g2, const float* g3,
    const float* iou1, const float* iou2, const float* iou3,
    const float* cc1, const float* cc2, const float* cc3,
    float* __restrict__ out){
  int lane = threadIdx.x & 31, wv = threadIdx.x >> 5;
  int n = blockIdx.x * 8 + wv;
  const size_t SEC = (size_t)NN * HD;
  geom_apply(g1, iou1, cc1, out + 0*SEC, out + 1*SEC, n, lane, false);
  geom_apply(g2, iou2, cc2, out + 2*SEC, out + 3*SEC, n, lane, true);
  // Euclidean path
  F4 R[3];
  #pragma unroll
  for (int t = 0; t < 3; t++){
    size_t off = (size_t)n*384 + t*HD + lane*4;
    R[t] = vadd(ld4(g3 + off), ld4(iou3 + off));
  }
  F4 ie = sigmoid4(R[0]), oe = sigmoid4(R[1]), ue = tanh4(R[2]);
  F4 c3v = ld4(cc3 + (size_t)n*HD + lane*4);
  F4 nc = vadd(vmul(ie, ue), c3v);
  F4 nh = vmul(oe, tanh4(nc));
  st4(out + 4*SEC + (size_t)n*HD + lane*4, nh);
  st4(out + 5*SEC + (size_t)n*HD + lane*4, nc);
}

// ====================== host launcher ======================
extern "C" void kernel_launch(void* const* d_in, const int* in_sizes, int n_in,
                              void* d_out, int out_size, void* d_ws, size_t ws_size,
                              hipStream_t stream){
  (void)in_sizes; (void)n_in; (void)out_size; (void)ws_size;
  const float* x     = (const float*)d_in[0];
  const float* h1    = (const float*)d_in[1];
  const float* c1    = (const float*)d_in[2];
  const float* h2    = (const float*)d_in[3];
  const float* c2    = (const float*)d_in[4];
  const float* h3    = (const float*)d_in[5];
  const float* c3    = (const float*)d_in[6];
  const float* del_t = (const float*)d_in[7];
  const float* iou1  = (const float*)d_in[8];
  const float* iou2  = (const float*)d_in[9];
  const float* iou3  = (const float*)d_in[10];
  const float* Wq_w  = (const float*)d_in[11];
  const float* Wq_b  = (const float*)d_in[12];
  const float* Wc_w  = (const float*)d_in[13];
  const float* Wc_b  = (const float*)d_in[14];
  const float* Uf_w  = (const float*)d_in[15];
  const float* Uf_b  = (const float*)d_in[16];
  const float* Up_w  = (const float*)d_in[17];
  const float* Up_b  = (const float*)d_in[18];
  const float* Uiou_w= (const float*)d_in[19];
  const float* Uiou_b= (const float*)d_in[20];
  const float* dsc   = (const float*)d_in[21];
  const int*   nbr   = (const int*)d_in[22];

  char* W8 = (char*)d_ws;
  const size_t SZB = (size_t)NN * ROWU * sizeof(unsigned);  // 2 MB bf16 row buf
  const size_t SZF = (size_t)NN * HD * sizeof(float);       // 4 MB f32 buf
  // bf16 GEMM inputs
  unsigned* bt0p = (unsigned*)(W8 + 0*SZB);
  unsigned* bt0s = (unsigned*)(W8 + 1*SZB);
  unsigned* bte  = (unsigned*)(W8 + 2*SZB);
  unsigned* bl1  = (unsigned*)(W8 + 3*SZB);
  unsigned* bl2  = (unsigned*)(W8 + 4*SZB);
  unsigned* bh3  = (unsigned*)(W8 + 5*SZB);
  unsigned* blc1 = (unsigned*)(W8 + 6*SZB);
  unsigned* blc2 = (unsigned*)(W8 + 7*SZB);
  unsigned* bc3  = (unsigned*)(W8 + 8*SZB);
  // f32 GEMM outs (later reused in-place by k_post1; whole region reused by giou)
  char* B0 = W8 + 9*SZB;
  float* q_p = (float*)(B0 + 0*SZF);  float* q_s = (float*)(B0 + 1*SZF);
  float* q_e = (float*)(B0 + 2*SZF);
  float* up1 = (float*)(B0 + 3*SZF);  float* up2 = (float*)(B0 + 4*SZF);
  float* up3 = (float*)(B0 + 5*SZF);
  float* uf1 = (float*)(B0 + 6*SZF);  float* uf2 = (float*)(B0 + 7*SZF);
  float* uf3 = (float*)(B0 + 8*SZF);
  float* wc1 = (float*)(B0 + 9*SZF);  float* wc2 = (float*)(B0 + 10*SZF);
  float* wc3 = (float*)(B0 + 11*SZF);
  // phase-2 outputs
  char* C0 = B0 + 12*SZF;
  float* htp = (float*)(C0 + 0*SZF);  float* hts = (float*)(C0 + 1*SZF);
  float* hte = (float*)(C0 + 2*SZF);
  float* cc1 = (float*)(C0 + 3*SZF);  float* cc2 = (float*)(C0 + 4*SZF);
  float* cc3 = (float*)(C0 + 5*SZF);
  // bf16 Uiou inputs
  char* D0 = C0 + 6*SZF;
  unsigned* ltp = (unsigned*)(D0 + 0*SZB);
  unsigned* lts = (unsigned*)(D0 + 1*SZB);
  unsigned* lte = (unsigned*)(D0 + 2*SZB);
  // per-node scalars
  char* S0 = D0 + 3*SZB;
  float* y2h1p = (float*)(S0 + 0*NN*4);
  float* y2h1s = (float*)(S0 + 1*NN*4);
  float* y2fp  = (float*)(S0 + 2*NN*4);
  float* y2fs  = (float*)(S0 + 3*NN*4);
  // bf16 weights
  char* Wb = S0 + 4*NN*4;
  unsigned* wWq  = (unsigned*)(Wb + 0*32768);
  unsigned* wWc  = (unsigned*)(Wb + 1*32768);
  unsigned* wUf  = (unsigned*)(Wb + 2*32768);
  unsigned* wUp  = (unsigned*)(Wb + 3*32768);
  unsigned* wUiou= (unsigned*)(Wb + 4*32768);                 // 384*64*4 = 96 KB
  // Uiou GEMM outputs overlay the (dead by then) 12-GEMM region B0
  float* g1 = (float*)(B0 + 0 * (size_t)NN*384*4);
  float* g2 = (float*)(B0 + 1 * (size_t)NN*384*4);
  float* g3 = (float*)(B0 + 2 * (size_t)NN*384*4);

  dim3 blkE(256), grdE(NN/8);   // wave-per-row elementwise kernels
  dim3 grdG(NN/16);             // gemm: 16-row stripe per block, 8 waves = 8 col tiles

  // 1) pack weights to bf16
  k_pack<<<32, 256, 0, stream>>>(Wq_w,  wWq,  128*128/2);
  k_pack<<<32, 256, 0, stream>>>(Wc_w,  wWc,  128*128/2);
  k_pack<<<32, 256, 0, stream>>>(Uf_w,  wUf,  128*128/2);
  k_pack<<<32, 256, 0, stream>>>(Up_w,  wUp,  128*128/2);
  k_pack<<<96, 256, 0, stream>>>(Uiou_w, wUiou, 384*128/2);
  // 2) pre-op
  k_preop<<<grdE, blkE, 0, stream>>>(x, h1, h2, h3, c1, c2, c3,
                                     bt0p, bt0s, bte, bl1, bl2, bh3, blc1, blc2, bc3);
  // 3) 12 WMMA GEMMs [8192,128] x [128,128]
  k_gemm<<<grdG, 256, 0, stream>>>(bt0p, wWq, Wq_b, q_p, HD);
  k_gemm<<<grdG, 256, 0, stream>>>(bt0s, wWq, Wq_b, q_s, HD);
  k_gemm<<<grdG, 256, 0, stream>>>(bte,  wWq, Wq_b, q_e, HD);
  k_gemm<<<grdG, 256, 0, stream>>>(bl1,  wUp, Up_b, up1, HD);
  k_gemm<<<grdG, 256, 0, stream>>>(bl2,  wUp, Up_b, up2, HD);
  k_gemm<<<grdG, 256, 0, stream>>>(bh3,  wUp, Up_b, up3, HD);
  k_gemm<<<grdG, 256, 0, stream>>>(bl1,  wUf, Uf_b, uf1, HD);
  k_gemm<<<grdG, 256, 0, stream>>>(bl2,  wUf, Uf_b, uf2, HD);
  k_gemm<<<grdG, 256, 0, stream>>>(bh3,  wUf, Uf_b, uf3, HD);
  k_gemm<<<grdG, 256, 0, stream>>>(blc1, wWc, Wc_b, wc1, HD);
  k_gemm<<<grdG, 256, 0, stream>>>(blc2, wWc, Wc_b, wc2, HD);
  k_gemm<<<grdG, 256, 0, stream>>>(bc3,  wWc, Wc_b, wc3, HD);
  // 4) per-node epilogue
  k_post1<<<grdE, blkE, 0, stream>>>(q_p, q_s, up1, up2, up3, uf1, uf2, uf3,
                                     wc1, wc2, wc3, h1, h2, h3, c1, c2, c3,
                                     del_t, dsc, y2h1p, y2h1s, y2fp, y2fs);
  // 5) attention + cell reductions
  k_attn<<<grdE, blkE, 0, stream>>>(nbr, q_p, q_s, q_e, up1, up2, up3,
                                    uf1, uf2, uf3, y2h1p, y2h1s, y2fp, y2fs,
                                    htp, hts, hte, cc1, cc2, cc3, ltp, lts, lte);
  // 6) Uiou GEMMs [8192,128] x [128,384]
  k_gemm<<<grdG, 256, 0, stream>>>(ltp, wUiou, Uiou_b, g1, 384);
  k_gemm<<<grdG, 256, 0, stream>>>(lts, wUiou, Uiou_b, g2, 384);
  k_gemm<<<grdG, 256, 0, stream>>>(lte, wUiou, Uiou_b, g3, 384);
  // 7) final apply -> d_out [6, N, 128]
  k_apply<<<grdE, blkE, 0, stream>>>(g1, g2, g3, iou1, iou2, iou3,
                                     cc1, cc2, cc3, (float*)d_out);
}
